// STGCNBlock_44521630991026
// MI455X (gfx1250) — matrix operand order repaired and checked by reading, not compiled, not run on hardware
//
#include <hip/hip_runtime.h>
#include <hip/hip_bf16.h>

// ---------------------------------------------------------------------------
// ST-GCN block, CDNA5 (gfx1250) wave32 WMMA implementation.
//
// B=64, C=64, T=512, V=14 (padded to 16). All channel-mixing math runs on
// v_wmma_f32_16x16x32_f16 (fp32 accumulate). The multi-scale depthwise conv +
// pointwise conv is folded into 13 shifted channel GEMMs with weights
// Wt[dt][o][c] = pw[o][c] * (dw5+dw9+dw13)[c][dt], so zero VALU conv work.
// LDS staging uses GLOBAL_LOAD_ASYNC_TO_LDS (ASYNCcnt) where available.
// ---------------------------------------------------------------------------

typedef _Float16 h16;
typedef __attribute__((ext_vector_type(16))) _Float16 v16h;
typedef __attribute__((ext_vector_type(8)))  _Float16 h8;
typedef __attribute__((ext_vector_type(8)))  float    v8f;
typedef __attribute__((ext_vector_type(4)))  int      v4i;

#define BB 64
#define CC 64
#define TT 512
#define VV 14
#define VP 16
#define TTILE 32
#define HALO 6
#define TRANGE (TTILE + 2*HALO)   // 44

// workspace byte offsets
#define WS_STATS   0        // 64*2 f32
#define WS_COLSUM  512      // 16 f32
#define WS_BNA     576      // 64 f32
#define WS_BNB     832      // 64 f32
#define WS_AFT     1088     // 256 f16 (AfT[w][v], padded 16x16)
#define WS_W1H     2048     // 64*64 f16
#define WS_WT      16384    // 13*64*64 f16
#define WS_G       131072   // 64*512*16*64 f16 = 64 MB

#if defined(__HIP_DEVICE_COMPILE__) && __has_builtin(__builtin_amdgcn_global_load_async_to_lds_b128)
#define USE_ASYNC 1
#else
#define USE_ASYNC 0
#endif

#if USE_ASYNC
__device__ __forceinline__ void async_g2l_b128(const void* gp, void* lp) {
  __builtin_amdgcn_global_load_async_to_lds_b128(
      (__attribute__((address_space(1))) v4i*)gp,
      (__attribute__((address_space(3))) v4i*)lp, 0, 0);
}
__device__ __forceinline__ void async_wait0() {
#if __has_builtin(__builtin_amdgcn_s_wait_asynccnt)
  __builtin_amdgcn_s_wait_asynccnt(0);
#else
  asm volatile("s_wait_asynccnt 0x0" ::: "memory");
#endif
}
#endif

__device__ __forceinline__ v16h mk16(h8 lo, h8 hi) {
  return __builtin_shufflevector(lo, hi, 0,1,2,3,4,5,6,7,8,9,10,11,12,13,14,15);
}
__device__ __forceinline__ v8f wmma16(v16h a, v16h b, v8f c) {
  // D = A(16x32 f16) * B(32x16 f16) + C(16x16 f32)
  return __builtin_amdgcn_wmma_f32_16x16x32_f16(false, a, false, b, (short)0, c, false, false);
}

// ---------------------------------------------------------------------------
// K0: tiny precompute: A_final, colsum, BN folds, f16 weights, fused Wt.
// ---------------------------------------------------------------------------
__global__ void k_prep(const float* __restrict__ A, const float* __restrict__ Bm,
                       const float* __restrict__ alpha_p,
                       const float* __restrict__ w1, const float* __restrict__ pw,
                       const float* __restrict__ dw5, const float* __restrict__ dw9,
                       const float* __restrict__ dw13,
                       const float* __restrict__ bn_g, const float* __restrict__ bn_b,
                       const float* __restrict__ bn_m, const float* __restrict__ bn_v,
                       h16* __restrict__ aft, float* __restrict__ colsum,
                       h16* __restrict__ w1h, h16* __restrict__ wt,
                       float* __restrict__ bnA, float* __restrict__ bnB) {
  __shared__ float At[VV][VV], An[VV][VV], Af[VV][VV], dis[VV];
  int tid = threadIdx.x;
  if (tid < VV*VV) {
    int i = tid / VV, j = tid % VV;
    At[i][j] = A[i*VV + j] + (i == j ? 1.f : 0.f);
  }
  __syncthreads();
  if (tid < VV) {
    float s = 0.f;
    for (int j = 0; j < VV; ++j) s += At[tid][j];
    s = s < 1.f ? 1.f : s;
    dis[tid] = rsqrtf(s);
  }
  __syncthreads();
  if (tid < VV*VV) {
    int i = tid / VV, j = tid % VV;
    An[i][j] = dis[i] * At[i][j] * dis[j];
  }
  if (tid < 256) aft[tid] = (h16)0.f;  // zero-pad 16x16 before valid writes
  __syncthreads();
  if (tid < VV*VV) {
    int i = tid / VV, j = tid % VV;
    float a2 = 0.f;
    for (int k = 0; k < VV; ++k) a2 += An[i][k] * An[k][j];
    float mx = -1e30f;
    for (int k = 0; k < VV; ++k) mx = fmaxf(mx, Bm[i*VV + k]);
    float se = 0.f;
    for (int k = 0; k < VV; ++k) se += __expf(Bm[i*VV + k] - mx);
    float bn = __expf(Bm[i*VV + j] - mx) / se;
    Af[i][j] = An[i][j] + 0.5f * a2 + alpha_p[0] * bn;
  }
  __syncthreads();
  if (tid < VV*VV) {
    int i = tid / VV, j = tid % VV;
    aft[j*VP + i] = (h16)Af[i][j];             // AfT[w][v]
  }
  if (tid < VP) {
    float cs = 0.f;
    if (tid < VV) for (int i = 0; i < VV; ++i) cs += Af[i][tid];
    colsum[tid] = cs;
  }
  if (tid < CC) {
    float a = bn_g[tid] * rsqrtf(bn_v[tid] + 1e-5f);
    bnA[tid] = a;
    bnB[tid] = bn_b[tid] - bn_m[tid] * a;
  }
  for (int i = tid; i < CC*CC; i += blockDim.x) w1h[i] = (h16)w1[i];
  for (int i = tid; i < 13*CC*CC; i += blockDim.x) {
    int dt = i >> 12, r = i & 4095, o = r >> 6, c = r & 63;
    float ks = dw13[c*13 + dt];
    if (dt >= 2 && dt <= 10) ks += dw9[c*9 + dt - 2];
    if (dt >= 4 && dt <= 8)  ks += dw5[c*5 + dt - 4];
    wt[i] = (h16)(pw[o*CC + c] * ks);
  }
}

// ---------------------------------------------------------------------------
// K1: per-batch GroupNorm(1,C) stats: mean + rsqrt(var+eps) over C*T*V.
// ---------------------------------------------------------------------------
__global__ __launch_bounds__(256) void k_stats(const float* __restrict__ x,
                                               float* __restrict__ stats) {
  const int b = blockIdx.x, tid = threadIdx.x;
  const int N4 = CC*TT*VV/4;  // 114688 float4s
  const float4* xb = (const float4*)(x + (size_t)b * CC*TT*VV);
  float s = 0.f, ss = 0.f;
  for (int i = tid; i < N4; i += 256) {
    float4 v = xb[i];
    s  += v.x + v.y + v.z + v.w;
    ss += v.x*v.x + v.y*v.y + v.z*v.z + v.w*v.w;
  }
  __shared__ float rs[256], rss[256];
  rs[tid] = s; rss[tid] = ss;
  __syncthreads();
  for (int off = 128; off; off >>= 1) {
    if (tid < off) { rs[tid] += rs[tid+off]; rss[tid] += rss[tid+off]; }
    __syncthreads();
  }
  if (tid == 0) {
    const float inv = 1.f / (float)(CC*TT*VV);
    float mu  = rs[0] * inv;
    float var = rss[0] * inv - mu*mu;
    stats[2*b]   = mu;
    stats[2*b+1] = rsqrtf(var + 1e-5f);
  }
}

// ---------------------------------------------------------------------------
// K3: GroupNorm apply -> vertex-agg WMMA (xn*Af) -> W1 GEMM WMMA -> relu -> g
//     g layout: [b][t][v(16)][c(64)] f16  (K-major for K4's B fragments)
// ---------------------------------------------------------------------------
__global__ __launch_bounds__(256) void k_gcn(const float* __restrict__ x,
                                             const float* __restrict__ ln_g,
                                             const float* __restrict__ ln_b,
                                             const float* __restrict__ b1,
                                             const float* __restrict__ stats,
                                             const float* __restrict__ colsum,
                                             const h16* __restrict__ aft,
                                             const h16* __restrict__ w1h,
                                             h16* __restrict__ g) {
  extern __shared__ char smem[];
  h16* xnL = (h16*)(smem);             // [32 t][64 c][16 v]  = 65536 B
  h16* yL  = (h16*)(smem + 65536);     // [512 n][64 c]       = 65536 B
  h16* w1L = (h16*)(smem + 131072);    // [64 o][64 c]        = 8192 B
  h16* afL = (h16*)(smem + 139264);    // AfT [16 w][16 v]    = 512 B

  const int b = blockIdx.x, t0 = blockIdx.y * TTILE;
  const int tid = threadIdx.x, ln = tid & 31;
  const int wv = __builtin_amdgcn_readfirstlane(tid >> 5);  // wave-uniform SGPR
  const int l2 = ln >> 4, lm = ln & 15;
  const float mu = stats[2*b], rstd = stats[2*b+1];

  // stage weights (async copy overlaps the GroupNorm VALU phase below)
#if USE_ASYNC
  for (int i8 = tid; i8 < CC*CC/8; i8 += 256)
    async_g2l_b128(&w1h[i8*8], &w1L[i8*8]);
  if (tid < 32) async_g2l_b128(&aft[tid*8], &afL[tid*8]);
#else
  for (int i = tid; i < CC*CC; i += 256) w1L[i] = w1h[i];
  if (tid < 256) afL[tid] = aft[tid];
#endif

  // GroupNorm apply -> xn f16 (v padded to 16 with zeros); dense float4 reads
  for (int i = tid; i < TTILE*CC/32; i += 256) {   // zero the v=14,15 pad lanes
    int t = i >> 6, c = i & 63;
    *(unsigned int*)&xnL[(t*CC + c)*VP + VV] = 0u;
  }
  for (int q = tid; q < CC*112; q += 256) {        // 112 float4s per channel row
    int c = q / 112, f = (q % 112) * 4;
    const float sc = rstd * ln_g[c], of = ln_b[c] - mu * rstd * ln_g[c];
    float4 xv = *(const float4*)&x[(((size_t)b*CC + c)*TT + t0)*VV + f];
    const float* xe = &xv.x;
#pragma unroll
    for (int e = 0; e < 4; ++e) {
      int ff = f + e, t = ff / VV, v = ff - t*VV;
      xnL[(t*CC + c)*VP + v] = (h16)(xe[e] * sc + of);
    }
  }
#if USE_ASYNC
  async_wait0();
#endif
  __syncthreads();

  // vertex aggregation: y[c, (t,w)] = sum_v xn[c,t,v] * Af[v,w]  (K=32, v>=14 zero)
  h8 z8 = {};
  v16h bAf;
  {
    h8 blo = z8, bhi = z8;
    {                                    // K=16..31 rows of padded Af are zero
      blo = *(const h8*)&afL[lm*VP + 8*l2];
    }
    bAf = mk16(blo, bhi);
  }
  for (int tile = wv; tile < 128; tile += 8) {     // scalar loop (wv uniform)
    int t = tile >> 2, ct = tile & 3;
    h8 alo = *(const h8*)&xnL[(t*CC + ct*16 + lm)*VP + 8*l2];
    v16h afr = mk16(alo, z8);            // K>=16 is zero
    v8f acc = {};
    acc = wmma16(afr, bAf, acc);
    h8 pk;
#pragma unroll
    for (int r = 0; r < 8; ++r) pk[r] = (h16)acc[r];
    *(h8*)&yL[(t*VP + lm)*CC + ct*16 + 8*l2] = pk;
  }
  __syncthreads();

  // GEMM1: g = relu(W1 * y + b1*colsum)
  const float cw = colsum[lm];
  for (int tile = wv; tile < 128; tile += 8) {     // scalar loop (wv uniform)
    int mt = tile >> 5, nt = tile & 31;
    v8f acc = {};
#pragma unroll
    for (int r = 0; r < 8; ++r) acc[r] = b1[mt*16 + 8*l2 + r] * cw;
#pragma unroll
    for (int kt = 0; kt < 2; ++kt) {
      const h16* wr = &w1L[(mt*16 + lm)*CC + kt*32];
      v16h afr = mk16(*(const h8*)&wr[8*l2], *(const h8*)&wr[16 + 8*l2]);
      const h16* yr = &yL[(nt*VP + lm)*CC + kt*32 + 8*l2];
      v16h bfr = mk16(*(const h8*)&yr[0], *(const h8*)&yr[16]);
      acc = wmma16(afr, bfr, acc);
    }
    h8 pk;
#pragma unroll
    for (int r = 0; r < 8; ++r) pk[r] = (h16)fmaxf(acc[r], 0.f);
    *(h8*)&g[(((size_t)b*TT + t0 + nt)*VP + lm)*CC + mt*16 + 8*l2] = pk;
  }
}

// ---------------------------------------------------------------------------
// K4: 13-tap fused (dw+pw) GEMM on WMMA + BN + relu + SE gate + residual.
// ---------------------------------------------------------------------------
__global__ __launch_bounds__(256) void k_tcn(const float* __restrict__ x,
                                             const h16* __restrict__ g,
                                             const h16* __restrict__ wt_g,
                                             const float* __restrict__ bnA,
                                             const float* __restrict__ bnB,
                                             const float* __restrict__ se1_w,
                                             const float* __restrict__ se1_b,
                                             const float* __restrict__ se2_w,
                                             const float* __restrict__ se2_b,
                                             float* __restrict__ out) {
  extern __shared__ char smem[];
  h16*   gL  = (h16*)(smem);              // [44 t][16 v][64 c] = 90112 B
  h16*   wtL = (h16*)(smem + 90112);      // [13 dt][64 o][64 c] = 106496 B
  h16*   o3L = (h16*)(smem + 196608);     // [32 t][16 v][64 c] = 65536 B
  float* at0 = (float*)(smem + 262144);   // [32 t][64 c]
  float* atn = (float*)(smem + 270336);   // [32 t][64 c]
  float* s1L = (float*)(smem + 278528);   // [32 t][8 m]

  const int b = blockIdx.x, t0 = blockIdx.y * TTILE;
  const int tid = threadIdx.x, ln = tid & 31;
  const int wv = __builtin_amdgcn_readfirstlane(tid >> 5);  // wave-uniform SGPR
  const int l2 = ln >> 4, lm = ln & 15;
  h8 z8 = {};

  // stage g tile with +/-6 halo (zero pad at sequence edges). In-range rows go
  // through the async memory->LDS path (ASYNCcnt); out-of-range rows are plain
  // DS zero stores at disjoint addresses (no async/DS ordering hazard).
  for (int i8 = tid; i8 < TRANGE*128; i8 += 256) {     // 128 h8-chunks per t-row
    int tl = i8 >> 7, rem = (i8 & 127) * 8;
    int gt = t0 - HALO + tl;
    if (gt >= 0 && gt < TT) {
#if USE_ASYNC
      async_g2l_b128(&g[((size_t)b*TT + gt)*(VP*CC) + rem], &gL[tl*(VP*CC) + rem]);
#else
      *(h8*)&gL[tl*(VP*CC) + rem] = *(const h8*)&g[((size_t)b*TT + gt)*(VP*CC) + rem];
#endif
    } else {
      *(h8*)&gL[tl*(VP*CC) + rem] = z8;
    }
  }
  for (int i8 = tid; i8 < 13*CC*CC/8; i8 += 256) {
#if USE_ASYNC
    async_g2l_b128(&wt_g[i8*8], &wtL[i8*8]);
#else
    *(h8*)&wtL[i8*8] = *(const h8*)&wt_g[i8*8];
#endif
  }
#if USE_ASYNC
  async_wait0();
#endif
  __syncthreads();

  // out2[o,(t,v)] = sum_dt sum_c Wt[dt][o][c] * g[c][t+dt-6][v]
  for (int tq = 0; tq < 4; ++tq) {
    const int tloc = wv*4 + tq;           // wave-uniform
    v8f acc[4];
#pragma unroll
    for (int mt = 0; mt < 4; ++mt) { v8f zz = {}; acc[mt] = zz; }
    for (int dt = 0; dt < 13; ++dt) {
#pragma unroll
      for (int kt = 0; kt < 2; ++kt) {
        const h16* gr = &gL[((tloc + dt)*VP + lm)*CC + kt*32 + 8*l2];
        v16h bfr = mk16(*(const h8*)&gr[0], *(const h8*)&gr[16]);
#pragma unroll
        for (int mt = 0; mt < 4; ++mt) {
          const h16* wr = &wtL[(dt*CC + mt*16 + lm)*CC + kt*32];
          v16h afr = mk16(*(const h8*)&wr[8*l2], *(const h8*)&wr[16 + 8*l2]);
          acc[mt] = wmma16(afr, bfr, acc[mt]);
        }
      }
    }
    // BN + relu -> o3 (f16 LDS)
#pragma unroll
    for (int mt = 0; mt < 4; ++mt) {
      h8 pk;
#pragma unroll
      for (int r = 0; r < 8; ++r) {
        int o = mt*16 + 8*l2 + r;
        float val = acc[mt][r] * bnA[o] + bnB[o];
        pk[r] = (h16)fmaxf(val, 0.f);
      }
      *(h8*)&o3L[(tloc*VP + lm)*CC + mt*16 + 8*l2] = pk;
    }
  }
  __syncthreads();

  // SE temporal gating: per (t): att0 = mean_v; s1 = relu(se1*att0+b); gate = sigmoid(se2*s1+b)
  const int tg = tid >> 3, lg = tid & 7;
#pragma unroll
  for (int j = 0; j < 8; ++j) {
    int o = lg*8 + j;
    float s = 0.f;
    for (int v = 0; v < VV; ++v) s += (float)o3L[(tg*VP + v)*CC + o];
    at0[tg*CC + o] = s * (1.f/14.f);
  }
  __syncthreads();
  {
    float s = se1_b[lg];
    for (int o = 0; o < CC; ++o) s += se1_w[lg*CC + o] * at0[tg*CC + o];
    s1L[tg*8 + lg] = fmaxf(s, 0.f);
  }
  __syncthreads();
#pragma unroll
  for (int j = 0; j < 8; ++j) {
    int o = lg*8 + j;
    float s = se2_b[o];
#pragma unroll
    for (int m = 0; m < 8; ++m) s += se2_w[o*8 + m] * s1L[tg*8 + m];
    atn[tg*CC + o] = 1.f / (1.f + __expf(-s));
  }
  __syncthreads();

  // gate * out + residual + relu -> d_out (f32, V=14 layout), float4 I/O
  for (int q = tid; q < CC*112; q += 256) {        // 112 float4s per channel row
    int c = q / 112, f = (q % 112) * 4;
    size_t gbase = (((size_t)b*CC + c)*TT + t0)*VV + f;
    float4 xv = *(const float4*)&x[gbase];
    float4 ov;
    const float* xe = &xv.x;
    float* oe = &ov.x;
#pragma unroll
    for (int e = 0; e < 4; ++e) {
      int ff = f + e, tl = ff / VV, v = ff - tl*VV;
      float o3 = (float)o3L[(tl*VP + v)*CC + c];
      oe[e] = fmaxf(o3 * atn[tl*CC + c] + xe[e], 0.f);
    }
    *(float4*)&out[gbase] = ov;
  }
}

// ---------------------------------------------------------------------------
extern "C" void kernel_launch(void* const* d_in, const int* in_sizes, int n_in,
                              void* d_out, int out_size, void* d_ws, size_t ws_size,
                              hipStream_t stream) {
  const float* x     = (const float*)d_in[0];
  const float* A     = (const float*)d_in[1];
  const float* ln_g  = (const float*)d_in[2];
  const float* ln_b  = (const float*)d_in[3];
  const float* w1    = (const float*)d_in[4];
  const float* b1    = (const float*)d_in[5];
  const float* Bm    = (const float*)d_in[6];
  const float* alpha = (const float*)d_in[7];
  const float* dw5   = (const float*)d_in[8];
  const float* dw9   = (const float*)d_in[9];
  const float* dw13  = (const float*)d_in[10];
  const float* pw    = (const float*)d_in[11];
  const float* bn_g  = (const float*)d_in[12];
  const float* bn_b  = (const float*)d_in[13];
  const float* bn_m  = (const float*)d_in[14];
  const float* bn_v  = (const float*)d_in[15];
  const float* se1_w = (const float*)d_in[16];
  const float* se1_b = (const float*)d_in[17];
  const float* se2_w = (const float*)d_in[18];
  const float* se2_b = (const float*)d_in[19];

  char* ws = (char*)d_ws;
  float* stats  = (float*)(ws + WS_STATS);
  float* colsum = (float*)(ws + WS_COLSUM);
  float* bnA    = (float*)(ws + WS_BNA);
  float* bnB    = (float*)(ws + WS_BNB);
  h16*   aft    = (h16*)(ws + WS_AFT);
  h16*   w1h    = (h16*)(ws + WS_W1H);
  h16*   wt     = (h16*)(ws + WS_WT);
  h16*   g      = (h16*)(ws + WS_G);

  const int shK3 = 65536 + 65536 + 8192 + 512;                     // 139776
  const int shK4 = 90112 + 106496 + 65536 + 8192 + 8192 + 1024;    // 279552
  (void)hipFuncSetAttribute((const void*)k_gcn,
      hipFuncAttributeMaxDynamicSharedMemorySize, shK3);
  (void)hipFuncSetAttribute((const void*)k_tcn,
      hipFuncAttributeMaxDynamicSharedMemorySize, shK4);

  k_prep<<<1, 256, 0, stream>>>(A, Bm, alpha, w1, pw, dw5, dw9, dw13,
                                bn_g, bn_b, bn_m, bn_v,
                                aft, colsum, w1h, wt, bnA, bnB);
  k_stats<<<BB, 256, 0, stream>>>(x, stats);
  k_gcn<<<dim3(BB, TT/TTILE), 256, shK3, stream>>>(x, ln_g, ln_b, b1, stats,
                                                   colsum, aft, w1h, g);
  k_tcn<<<dim3(BB, TT/TTILE), 256, shK4, stream>>>(x, g, wt, bnA, bnB,
                                                   se1_w, se1_b, se2_w, se2_b,
                                                   (float*)d_out);
}
